// GRU_17051020165453
// MI455X (gfx1250) — matrix operand order repaired and checked by reading, not compiled, not run on hardware
//
#include <hip/hip_runtime.h>
#include <hip/hip_bf16.h>

// ---------------------------------------------------------------------------
// GRU recommender loss on MI455X (gfx1250, wave32, WMMA f32<-f16 16x16x32)
//   K1: embedding gather + 50-step GRU scan  -> user_eb [256,64]
//   K2: fused user_eb @ emb_table.T (K=64) + sum-exp partials (4 rowtiles/blk)
//   K3: sum partials -> log; subtract target logit; mean -> scalar loss
// Logits are bounded (|h|<=1, emb ~ N(0,0.02) => |logit| <~ 6), so plain
// sum-of-exp is numerically identical to max-shifted log-softmax in fp32.
// ---------------------------------------------------------------------------

#define BATCH   256
#define SEQLEN  50
#define EMB     64
#define HID     64
#define GATES   192          // 3*HID
#define N_IID   100000
#define N_TILES 6250         // N_IID / 16
#define TPC     32           // 16-wide N-tiles per chunk
#define NCHUNK  196          // ceil(6250 / 32)
#define RT_BLK  4            // row-tiles (of 16 batch rows) per block in K2

typedef _Float16 v16h __attribute__((ext_vector_type(16)));
typedef _Float16 v4h  __attribute__((ext_vector_type(4)));
typedef float    v8f  __attribute__((ext_vector_type(8)));

// ---- A-fragment K index for 16-bit 16x32 A (contiguous: k = 16*half + e) ---
__device__ __forceinline__ int a_kidx(int e, int half) {
    return 16 * half + e;
}

__device__ __forceinline__ float fast_rcp(float x) {
    return __builtin_amdgcn_rcpf(x);            // v_rcp_f32, no div fixup
}
__device__ __forceinline__ float fast_sigmoid(float x) {
    return fast_rcp(1.0f + __expf(-x));
}
__device__ __forceinline__ float fast_tanh(float x) {
    const float e = __expf(2.0f * x);           // inf-safe: ->1 / ->-1
    return 1.0f - 2.0f * fast_rcp(e + 1.0f);
}

// ---------------------------------------------------------------------------
// Kernel 1: GRU over history. grid.x = 16 (16 batch rows each), block = 384
// (12 waves; wave w owns gate columns [16w,16w+16) and keeps its W_ih/W_hh
// WMMA B-fragments in registers for the whole 50-step scan).
// ---------------------------------------------------------------------------
__global__ __launch_bounds__(384)
void gru_scan_kernel(const int* __restrict__ his, const int* __restrict__ mask,
                     const float* __restrict__ emb,
                     const float* __restrict__ w_ih, const float* __restrict__ w_hh,
                     const float* __restrict__ b_ih, const float* __restrict__ b_hh,
                     float* __restrict__ user_f32, _Float16* __restrict__ user_f16)
{
    __shared__ _Float16 x16[16][EMB];   // current step input, f16
    __shared__ _Float16 h16[16][HID];   // hidden, f16 (WMMA operand)
    __shared__ float    h32[16][HID];   // hidden, f32 (exact update)
    __shared__ float    gi[16][GATES];  // x @ W_ih.T + b_ih
    __shared__ float    gh[16][GATES];  // h @ W_hh.T + b_hh

    const int tid  = threadIdx.x;
    const int w    = tid >> 5;          // wave 0..11
    const int lane = tid & 31;
    const int half = lane >> 4;
    const int l16  = lane & 15;
    const int b0   = blockIdx.x * 16;

    // ---- preload weight B-fragments (B[k][n] = W[n][k]), float4 loads ------
    const int col = w * 16 + l16;       // gate column 0..191 for this lane
    v16h Bi[2], Bh[2];
    #pragma unroll
    for (int s = 0; s < 2; ++s) {
        const int k0 = 32 * s + 16 * half;
        const float4* wi4 = reinterpret_cast<const float4*>(w_ih + col * EMB + k0);
        const float4* wh4 = reinterpret_cast<const float4*>(w_hh + col * HID + k0);
        #pragma unroll
        for (int q = 0; q < 4; ++q) {
            const float4 a = wi4[q], b = wh4[q];
            Bi[s][4 * q + 0] = (_Float16)a.x;  Bi[s][4 * q + 1] = (_Float16)a.y;
            Bi[s][4 * q + 2] = (_Float16)a.z;  Bi[s][4 * q + 3] = (_Float16)a.w;
            Bh[s][4 * q + 0] = (_Float16)b.x;  Bh[s][4 * q + 1] = (_Float16)b.y;
            Bh[s][4 * q + 2] = (_Float16)b.z;  Bh[s][4 * q + 3] = (_Float16)b.w;
        }
    }
    const float bias_i = b_ih[col];
    const float bias_h = b_hh[col];

    // ---- h = 0 -------------------------------------------------------------
    for (int idx = tid; idx < 16 * HID; idx += 384) {
        h32[idx >> 6][idx & 63] = 0.0f;
        h16[idx >> 6][idx & 63] = (_Float16)0.0f;
    }
    __syncthreads();

    const float4* emb4 = reinterpret_cast<const float4*>(emb);

    for (int t = 0; t < SEQLEN; ++t) {
        // stage x_t = emb_table[his[b,t]] * mask[b,t]  (f16, float4 granules)
        for (int idx = tid; idx < 16 * 16; idx += 384) {
            const int m = idx >> 4, q = idx & 15;
            const int b = b0 + m;
            const int item = his[b * SEQLEN + t];
            const float mv = (float)mask[b * SEQLEN + t];
            const float4 v = emb4[item * 16 + q];
            v4h hv = { (_Float16)(v.x * mv), (_Float16)(v.y * mv),
                       (_Float16)(v.z * mv), (_Float16)(v.w * mv) };
            *reinterpret_cast<v4h*>(&x16[m][q * 4]) = hv;
        }
        __syncthreads();

        // build A fragments (shared across waves -> read from LDS)
        v16h ax0, ax1, ah0, ah1;
        #pragma unroll
        for (int e = 0; e < 16; ++e) {
            const int kk = a_kidx(e, half);
            ax0[e] = x16[l16][kk];
            ax1[e] = x16[l16][32 + kk];
            ah0[e] = h16[l16][kk];
            ah1[e] = h16[l16][32 + kk];
        }
        v8f acc_i = {}, acc_h = {};
        acc_i = __builtin_amdgcn_wmma_f32_16x16x32_f16(false, ax0, false, Bi[0], (short)0, acc_i, false, false);
        acc_i = __builtin_amdgcn_wmma_f32_16x16x32_f16(false, ax1, false, Bi[1], (short)0, acc_i, false, false);
        acc_h = __builtin_amdgcn_wmma_f32_16x16x32_f16(false, ah0, false, Bh[0], (short)0, acc_h, false, false);
        acc_h = __builtin_amdgcn_wmma_f32_16x16x32_f16(false, ah1, false, Bh[1], (short)0, acc_h, false, false);

        #pragma unroll
        for (int r = 0; r < 8; ++r) {
            const int m = r + 8 * half;   // C/D layout: row = r + 8*(lane/16)
            gi[m][col] = acc_i[r] + bias_i;
            gh[m][col] = acc_h[r] + bias_h;
        }
        __syncthreads();

        // gate nonlinearities + hidden update (torch GRU gate order r,z,n)
        for (int idx = tid; idx < 16 * HID; idx += 384) {
            const int m = idx >> 6, j = idx & 63;
            const float r  = fast_sigmoid(gi[m][j]       + gh[m][j]);
            const float z  = fast_sigmoid(gi[m][64 + j]  + gh[m][64 + j]);
            const float nn = fast_tanh   (gi[m][128 + j] + r * gh[m][128 + j]);
            const float hn = (1.0f - z) * nn + z * h32[m][j];
            h32[m][j] = hn;
            h16[m][j] = (_Float16)hn;
        }
        __syncthreads();
    }

    for (int idx = tid; idx < 16 * HID; idx += 384) {
        const int m = idx >> 6, k = idx & 63;
        user_f32[(b0 + m) * HID + k] = h32[m][k];
        user_f16[(b0 + m) * HID + k] = h16[m][k];
    }
}

// ---------------------------------------------------------------------------
// Kernel 2: rank tile GEMM fused with sum-exp partials.
// grid = (NCHUNK, 4), block = 256 (8 waves). Each block covers 64 batch rows
// (4 row-tiles) x one 512-item chunk; wave w owns N-tiles {chunk*32 + w + 8i}.
// One B-fragment fetch feeds 8 WMMAs. Logits never touch memory: folded into
// per-lane sum-exp, reduced via width-16 shuffles then LDS across waves.
// ---------------------------------------------------------------------------
__global__ __launch_bounds__(256)
void rank_softmax_partial(const float* __restrict__ emb,
                          const _Float16* __restrict__ user_f16,
                          float* __restrict__ psum)
{
    __shared__ float wsum[8][RT_BLK * 16];

    const int tid  = threadIdx.x;
    const int w    = tid >> 5;
    const int lane = tid & 31;
    const int half = lane >> 4;
    const int l16  = lane & 15;
    const int chunk = blockIdx.x;
    const int rb    = blockIdx.y;       // row-block: rows [rb*64, rb*64+64)

    // A fragments for 4 row-tiles (K=64 -> 2 frags each), loaded once
    v16h A[RT_BLK][2];
    #pragma unroll
    for (int j = 0; j < RT_BLK; ++j) {
        const int arow = (rb * RT_BLK + j) * 16 + l16;
        #pragma unroll
        for (int e = 0; e < 16; ++e) {
            const int kk = a_kidx(e, half);
            A[j][0][e] = user_f16[arow * HID + kk];
            A[j][1][e] = user_f16[arow * HID + 32 + kk];
        }
    }

    float rs[RT_BLK][8];
    #pragma unroll
    for (int j = 0; j < RT_BLK; ++j)
        #pragma unroll
        for (int r = 0; r < 8; ++r) rs[j][r] = 0.0f;

    const int tbase = chunk * TPC;
    for (int i = 0; i < 4; ++i) {
        const int tile = tbase + w + 8 * i;         // wave-uniform
        if (tile >= N_TILES) break;
        const int item = tile * 16 + l16;           // B column = item row

        if (tile + 8 < N_TILES) {                   // prefetch next tile's rows
            const int nitem = (tile + 8) * 16 + l16;
            __builtin_prefetch(emb + nitem * EMB, 0, 3);
        }

        // B fragments: 16 consecutive floats per k-half, float4 + cvt
        const float4* p = reinterpret_cast<const float4*>(emb + item * EMB + 16 * half);
        v16h bb0, bb1;
        #pragma unroll
        for (int q = 0; q < 4; ++q) {
            const float4 v = p[q];                  // K = 16*half + 4q + ...
            const float4 u = p[q + 8];              // K = 32 + 16*half + 4q + ...
            bb0[4 * q + 0] = (_Float16)v.x;  bb0[4 * q + 1] = (_Float16)v.y;
            bb0[4 * q + 2] = (_Float16)v.z;  bb0[4 * q + 3] = (_Float16)v.w;
            bb1[4 * q + 0] = (_Float16)u.x;  bb1[4 * q + 1] = (_Float16)u.y;
            bb1[4 * q + 2] = (_Float16)u.z;  bb1[4 * q + 3] = (_Float16)u.w;
        }

        #pragma unroll
        for (int j = 0; j < RT_BLK; ++j) {
            v8f acc = {};
            acc = __builtin_amdgcn_wmma_f32_16x16x32_f16(false, A[j][0], false, bb0, (short)0, acc, false, false);
            acc = __builtin_amdgcn_wmma_f32_16x16x32_f16(false, A[j][1], false, bb1, (short)0, acc, false, false);
            #pragma unroll
            for (int r = 0; r < 8; ++r)
                rs[j][r] += __expf(acc[r]);         // bounded logits: safe
        }
    }

    // sum across the 16 lanes that share each row (width-16 xor shuffles)
    #pragma unroll
    for (int off = 1; off < 16; off <<= 1)
        #pragma unroll
        for (int j = 0; j < RT_BLK; ++j)
            #pragma unroll
            for (int r = 0; r < 8; ++r)
                rs[j][r] += __shfl_xor(rs[j][r], off, 16);

    if (l16 == 0) {
        #pragma unroll
        for (int j = 0; j < RT_BLK; ++j)
            #pragma unroll
            for (int r = 0; r < 8; ++r)
                wsum[w][j * 16 + 8 * half + r] = rs[j][r];
    }
    __syncthreads();

    if (tid < RT_BLK * 16) {                        // combine 8 waves -> chunk
        float S = 0.0f;
        #pragma unroll
        for (int ww = 0; ww < 8; ++ww) S += wsum[ww][tid];
        const int row = rb * (RT_BLK * 16) + tid;
        psum[row * NCHUNK + chunk] = S;
    }
}

// ---------------------------------------------------------------------------
// Kernel 3: sum chunk partials -> log; subtract target logit; mean.
// ---------------------------------------------------------------------------
__global__ __launch_bounds__(256)
void final_loss(const int* __restrict__ targets, const float* __restrict__ emb,
                const float* __restrict__ user_f32,
                const float* __restrict__ psum, float* __restrict__ out)
{
    __shared__ float red[BATCH];
    const int b = threadIdx.x;

    float S = 0.0f;
    for (int c = 0; c < NCHUNK; ++c) S += psum[b * NCHUNK + c];
    const float lse = __logf(S);

    const int tgt = targets[b];
    float dot = 0.0f;
    #pragma unroll 8
    for (int k = 0; k < EMB; ++k)
        dot += user_f32[b * EMB + k] * emb[tgt * EMB + k];

    red[b] = lse - dot;                 // -log p(target)
    __syncthreads();
    for (int s = BATCH / 2; s > 0; s >>= 1) {
        if (b < s) red[b] += red[b + s];
        __syncthreads();
    }
    if (b == 0) out[0] = red[0] / (float)BATCH;
}

// ---------------------------------------------------------------------------
extern "C" void kernel_launch(void* const* d_in, const int* in_sizes, int n_in,
                              void* d_out, int out_size, void* d_ws, size_t ws_size,
                              hipStream_t stream)
{
    const int*   mid   = (const int*)  d_in[0];   // [256] targets
    const int*   his   = (const int*)  d_in[1];   // [256,50]
    const int*   mask  = (const int*)  d_in[2];   // [256,50]
    /* d_in[3] = neg_item: unused on the full-softmax loss path */
    const float* emb   = (const float*)d_in[4];   // [100000,64]
    const float* w_ih  = (const float*)d_in[5];   // [192,64]
    const float* w_hh  = (const float*)d_in[6];   // [192,64]
    const float* b_ih  = (const float*)d_in[7];   // [192]
    const float* b_hh  = (const float*)d_in[8];   // [192]

    char* ws = (char*)d_ws;
    float*    user_f32 = (float*)   (ws + 0);        // 256*64*4 = 65536 B
    _Float16* user_f16 = (_Float16*)(ws + 65536);    // 256*64*2 = 32768 B
    float*    psum     = (float*)   (ws + 98304);    // 256*196*4 = 200704 B

    gru_scan_kernel<<<16, 384, 0, stream>>>(his, mask, emb, w_ih, w_hh,
                                            b_ih, b_hh, user_f32, user_f16);
    rank_softmax_partial<<<dim3(NCHUNK, RT_BLK), 256, 0, stream>>>(emb, user_f16,
                                                                   psum);
    final_loss<<<1, 256, 0, stream>>>(mid, emb, user_f32, psum, (float*)d_out);
}
